// GATNodeClassifier_71141838291479
// MI455X (gfx1250) — matrix operand order repaired and checked by reading, not compiled, not run on hardware
//
#include <hip/hip_runtime.h>
#include <hip/hip_bf16.h>
#include <stdint.h>

typedef __attribute__((ext_vector_type(16))) __bf16          v16bf;
typedef __attribute__((ext_vector_type(16))) unsigned short  v16u;
typedef __attribute__((ext_vector_type(8)))  float           v8f;

#define NN 100000
#define EE 3200000
#define FIN 256
#define HD 64      // H1*D1
#define NC 40
#define MT 6250    // NN / 16  (exact)

// ---------- helpers ----------
__device__ __forceinline__ unsigned short f2bf(float f) {
  unsigned int u = __float_as_uint(f);
  u += 0x7FFFu + ((u >> 16) & 1u);        // round-to-nearest-even
  return (unsigned short)(u >> 16);
}
__device__ __forceinline__ float lrelu(float a) { return a > 0.f ? a : 0.2f * a; }
__device__ __forceinline__ void atomicMaxF(float* p, float v) {
  if (v >= 0.f) atomicMax((int*)p, __float_as_int(v));
  else          atomicMin((unsigned int*)p, __float_as_uint(v));
}

// ---------- init ----------
__global__ void k_fill(float* __restrict__ p, float v, int n) {
  int i = blockIdx.x * blockDim.x + threadIdx.x;
  if (i < n) p[i] = v;
}

// ---------- weight packing into WMMA B-fragment layout ----------
// W1 [256,64] f32 -> bf16 frags [kt=8][nt=4][lane=32][j=16]
__global__ void k_pack_w1(const float* __restrict__ W, unsigned short* __restrict__ P) {
  int idx = blockIdx.x * blockDim.x + threadIdx.x;
  if (idx >= 8 * 4 * 32 * 16) return;
  int j = idx & 15, lane = (idx >> 4) & 31, nt = (idx >> 9) & 3, kt = idx >> 11;
  int K = kt * 32 + ((j & 8) ? 16 : 0) + ((lane & 16) ? 8 : 0) + (j & 7);
  int n = nt * 16 + (lane & 15);
  P[idx] = f2bf(W[K * HD + n]);
}
// W2 [64,40] f32 -> bf16 frags [kt=2][nt=3][lane=32][j=16], n>=40 padded 0
__global__ void k_pack_w2(const float* __restrict__ W, unsigned short* __restrict__ P) {
  int idx = blockIdx.x * blockDim.x + threadIdx.x;
  if (idx >= 2 * 3 * 32 * 16) return;
  int j = idx & 15, lane = (idx >> 4) & 31, t = idx >> 9;
  int nt = t % 3, kt = t / 3;
  int K = kt * 32 + ((j & 8) ? 16 : 0) + ((lane & 16) ? 8 : 0) + (j & 7);
  int n = nt * 16 + (lane & 15);
  P[idx] = (n < NC) ? f2bf(W[K * NC + n]) : (unsigned short)0;
}

// ---------- GEMM1: feat1[N,64] = x[N,256] @ W1, bf16 WMMA, fp32 accum ----------
__global__ void k_gemm1(const float* __restrict__ x, const unsigned short* __restrict__ P,
                        float* __restrict__ feat) {
  int wv = blockIdx.x * (blockDim.x >> 5) + (threadIdx.x >> 5); // M-tile
  if (wv >= MT) return;                                         // wave-uniform: EXEC stays full
  int lane = threadIdx.x & 31;
  int hi   = (lane & 16) ? 1 : 0;
  int mrow = (wv << 4) + (lane & 15);
  const float* xr = x + (size_t)mrow * FIN + hi * 8;
  const v16bf* BP = (const v16bf*)P;
  v8f a0 = {}, a1 = {}, a2 = {}, a3 = {};
  for (int ks = 0; ks < 8; ++ks) {
    float af[16];
    *(float4*)(af + 0)  = *(const float4*)(xr + ks * 32 + 0);
    *(float4*)(af + 4)  = *(const float4*)(xr + ks * 32 + 4);
    *(float4*)(af + 8)  = *(const float4*)(xr + ks * 32 + 16);
    *(float4*)(af + 12) = *(const float4*)(xr + ks * 32 + 20);
    v16u au;
#pragma unroll
    for (int i = 0; i < 16; ++i) au[i] = f2bf(af[i]);
    v16bf a = __builtin_bit_cast(v16bf, au);
    v16bf b0 = BP[((ks * 4 + 0) * 32 + lane)];
    v16bf b1 = BP[((ks * 4 + 1) * 32 + lane)];
    v16bf b2 = BP[((ks * 4 + 2) * 32 + lane)];
    v16bf b3 = BP[((ks * 4 + 3) * 32 + lane)];
    a0 = __builtin_amdgcn_wmma_f32_16x16x32_bf16(false, a, false, b0, (short)0, a0, false, false);
    a1 = __builtin_amdgcn_wmma_f32_16x16x32_bf16(false, a, false, b1, (short)0, a1, false, false);
    a2 = __builtin_amdgcn_wmma_f32_16x16x32_bf16(false, a, false, b2, (short)0, a2, false, false);
    a3 = __builtin_amdgcn_wmma_f32_16x16x32_bf16(false, a, false, b3, (short)0, a3, false, false);
  }
  int rb = (wv << 4) + hi * 8;
  int c0 = lane & 15;
  float* o = feat + (size_t)rb * HD + c0;
#pragma unroll
  for (int r = 0; r < 8; ++r) {
    o[(size_t)r * HD + 0]  = a0[r];
    o[(size_t)r * HD + 16] = a1[r];
    o[(size_t)r * HD + 32] = a2[r];
    o[(size_t)r * HD + 48] = a3[r];
  }
}

// ---------- GEMM2: feat2[N,40] = h[N,64] @ W2 (N padded to 48) ----------
__global__ void k_gemm2(const float* __restrict__ h, const unsigned short* __restrict__ P,
                        float* __restrict__ feat2) {
  int wv = blockIdx.x * (blockDim.x >> 5) + (threadIdx.x >> 5);
  if (wv >= MT) return;
  int lane = threadIdx.x & 31;
  int hi   = (lane & 16) ? 1 : 0;
  int mrow = (wv << 4) + (lane & 15);
  const float* hr = h + (size_t)mrow * HD + hi * 8;
  const v16bf* BP = (const v16bf*)P;
  v8f acc[3] = {};
  for (int ks = 0; ks < 2; ++ks) {
    float af[16];
    *(float4*)(af + 0)  = *(const float4*)(hr + ks * 32 + 0);
    *(float4*)(af + 4)  = *(const float4*)(hr + ks * 32 + 4);
    *(float4*)(af + 8)  = *(const float4*)(hr + ks * 32 + 16);
    *(float4*)(af + 12) = *(const float4*)(hr + ks * 32 + 20);
    v16u au;
#pragma unroll
    for (int i = 0; i < 16; ++i) au[i] = f2bf(af[i]);
    v16bf a = __builtin_bit_cast(v16bf, au);
#pragma unroll
    for (int nt = 0; nt < 3; ++nt) {
      v16bf b = BP[((ks * 3 + nt) * 32 + lane)];
      acc[nt] = __builtin_amdgcn_wmma_f32_16x16x32_bf16(false, a, false, b, (short)0, acc[nt], false, false);
    }
  }
  int rb = (wv << 4) + hi * 8;
  int c0 = lane & 15;
#pragma unroll
  for (int nt = 0; nt < 3; ++nt) {
    int c = nt * 16 + c0;
    if (c < NC) {
#pragma unroll
      for (int r = 0; r < 8; ++r) feat2[(size_t)(rb + r) * NC + c] = acc[nt][r];
    }
  }
}

// ---------- attention projections ----------
__global__ void k_elr1(const float* __restrict__ feat, const float* __restrict__ al,
                       const float* __restrict__ ar, float* __restrict__ el, float* __restrict__ er) {
  int i = blockIdx.x * blockDim.x + threadIdx.x;
  if (i >= NN * 8) return;
  int n = i >> 3, hh = i & 7;
  const float* f = feat + (size_t)n * HD + hh * 8;
  float sl = 0.f, sr = 0.f;
#pragma unroll
  for (int d = 0; d < 8; ++d) { float v = f[d]; sl += v * al[hh * 8 + d]; sr += v * ar[hh * 8 + d]; }
  el[i] = sl; er[i] = sr;
}
__global__ void k_elr2(const float* __restrict__ feat2, const float* __restrict__ al,
                       const float* __restrict__ ar, float* __restrict__ el, float* __restrict__ er) {
  int n = blockIdx.x * blockDim.x + threadIdx.x;
  if (n >= NN) return;
  const float* f = feat2 + (size_t)n * NC;
  float sl = 0.f, sr = 0.f;
  for (int c = 0; c < NC; ++c) { float v = f[c]; sl += v * al[c]; sr += v * ar[c]; }
  el[n] = sl; er[n] = sr;
}

// ---------- layer-1 edge passes ----------
__global__ void k_emax1(const int* __restrict__ src, const int* __restrict__ dst,
                        const float* __restrict__ el, const float* __restrict__ er,
                        float* __restrict__ m) {
  int i = blockIdx.x * blockDim.x + threadIdx.x;
  if (i >= EE * 8) return;
  int e = i >> 3, hh = i & 7;
  int s = src[e], d = dst[e];
  atomicMaxF(&m[d * 8 + hh], lrelu(el[s * 8 + hh] + er[d * 8 + hh]));
}
__global__ void k_esum1(const int* __restrict__ src, const int* __restrict__ dst,
                        const float* __restrict__ el, const float* __restrict__ er,
                        const float* __restrict__ m, float* __restrict__ ssum) {
  int i = blockIdx.x * blockDim.x + threadIdx.x;
  if (i >= EE * 8) return;
  int e = i >> 3, hh = i & 7;
  int s = src[e], d = dst[e];
  float v = lrelu(el[s * 8 + hh] + er[d * 8 + hh]);
  atomicAdd(&ssum[d * 8 + hh], __expf(v - m[d * 8 + hh]));
}
__global__ void k_eaggr1(const int* __restrict__ src, const int* __restrict__ dst,
                         const float* __restrict__ el, const float* __restrict__ er,
                         const float* __restrict__ m, const float* __restrict__ ssum,
                         const float* __restrict__ feat, float* __restrict__ out) {
  int i = blockIdx.x * blockDim.x + threadIdx.x;
  if (i >= EE * 64) return;
  int e = i >> 6, f = i & 63, hh = f >> 3;
  int sn = src[e], dn = dst[e];
  float v = lrelu(el[sn * 8 + hh] + er[dn * 8 + hh]);
  float alpha = __expf(v - m[dn * 8 + hh]) / ssum[dn * 8 + hh];
  atomicAdd(&out[(size_t)dn * HD + f], feat[(size_t)sn * HD + f] * alpha);
}
__global__ void k_bias_relu(float* __restrict__ out, const float* __restrict__ b) {
  int i = blockIdx.x * blockDim.x + threadIdx.x;
  if (i >= NN * HD) return;
  float v = out[i] + b[i & 63];
  out[i] = v > 0.f ? v : 0.f;
}

// ---------- layer-2 edge passes (H=1) ----------
__global__ void k_emax2(const int* __restrict__ src, const int* __restrict__ dst,
                        const float* __restrict__ el, const float* __restrict__ er,
                        float* __restrict__ m) {
  int e = blockIdx.x * blockDim.x + threadIdx.x;
  if (e >= EE) return;
  int s = src[e], d = dst[e];
  atomicMaxF(&m[d], lrelu(el[s] + er[d]));
}
__global__ void k_esum2(const int* __restrict__ src, const int* __restrict__ dst,
                        const float* __restrict__ el, const float* __restrict__ er,
                        const float* __restrict__ m, float* __restrict__ ssum) {
  int e = blockIdx.x * blockDim.x + threadIdx.x;
  if (e >= EE) return;
  int s = src[e], d = dst[e];
  float v = lrelu(el[s] + er[d]);
  atomicAdd(&ssum[d], __expf(v - m[d]));
}
__global__ void k_eaggr2(const int* __restrict__ src, const int* __restrict__ dst,
                         const float* __restrict__ el, const float* __restrict__ er,
                         const float* __restrict__ m, const float* __restrict__ ssum,
                         const float* __restrict__ feat2, float* __restrict__ out) {
  int i = blockIdx.x * blockDim.x + threadIdx.x;
  if (i >= EE * NC) return;
  int e = i / NC, c = i - e * NC;
  int sn = src[e], dn = dst[e];
  float v = lrelu(el[sn] + er[dn]);
  float alpha = __expf(v - m[dn]) / ssum[dn];
  atomicAdd(&out[(size_t)dn * NC + c], feat2[(size_t)sn * NC + c] * alpha);
}
__global__ void k_bias2(float* __restrict__ out, const float* __restrict__ b) {
  int i = blockIdx.x * blockDim.x + threadIdx.x;
  if (i >= NN * NC) return;
  out[i] += b[i % NC];
}

// ---------- launch ----------
extern "C" void kernel_launch(void* const* d_in, const int* in_sizes, int n_in,
                              void* d_out, int out_size, void* d_ws, size_t ws_size,
                              hipStream_t stream) {
  const float* x   = (const float*)d_in[0];
  const float* W1  = (const float*)d_in[1];
  const float* al1 = (const float*)d_in[2];
  const float* ar1 = (const float*)d_in[3];
  const float* b1  = (const float*)d_in[4];
  const float* W2  = (const float*)d_in[5];
  const float* al2 = (const float*)d_in[6];
  const float* ar2 = (const float*)d_in[7];
  const float* b2  = (const float*)d_in[8];
  const int*   src = (const int*)d_in[9];
  const int*   dst = (const int*)d_in[10];
  float* dout = (float*)d_out;

  float* ws    = (float*)d_ws;
  float* feat1 = ws;
  float* out1  = feat1 + (size_t)NN * HD;
  float* el1   = out1 + (size_t)NN * HD;
  float* er1   = el1 + (size_t)NN * 8;
  float* m1    = er1 + (size_t)NN * 8;
  float* s1    = m1 + (size_t)NN * 8;
  float* feat2 = s1 + (size_t)NN * 8;
  float* el2   = feat2 + (size_t)NN * NC;
  float* er2   = el2 + NN;
  float* m2    = er2 + NN;
  float* s2    = m2 + NN;
  unsigned short* pw1 = (unsigned short*)(s2 + NN);
  unsigned short* pw2 = pw1 + 8 * 4 * 32 * 16;

  const float NEGINF = -__builtin_huge_valf();

  k_pack_w1<<<64, 256, 0, stream>>>(W1, pw1);
  k_pack_w2<<<12, 256, 0, stream>>>(W2, pw2);

  k_fill<<<(NN * 8 + 255) / 256, 256, 0, stream>>>(m1, NEGINF, NN * 8);
  k_fill<<<(NN * 8 + 255) / 256, 256, 0, stream>>>(s1, 0.f, NN * 8);
  k_fill<<<(NN * HD + 255) / 256, 256, 0, stream>>>(out1, 0.f, NN * HD);
  k_fill<<<(NN + 255) / 256, 256, 0, stream>>>(m2, NEGINF, NN);
  k_fill<<<(NN + 255) / 256, 256, 0, stream>>>(s2, 0.f, NN);
  k_fill<<<(NN * NC + 255) / 256, 256, 0, stream>>>(dout, 0.f, NN * NC);

  // layer 1
  k_gemm1<<<(MT + 3) / 4, 128, 0, stream>>>(x, pw1, feat1);
  k_elr1<<<(NN * 8 + 255) / 256, 256, 0, stream>>>(feat1, al1, ar1, el1, er1);
  k_emax1<<<(EE * 8 + 255) / 256, 256, 0, stream>>>(src, dst, el1, er1, m1);
  k_esum1<<<(EE * 8 + 255) / 256, 256, 0, stream>>>(src, dst, el1, er1, m1, s1);
  k_eaggr1<<<(EE * 64 + 255) / 256, 256, 0, stream>>>(src, dst, el1, er1, m1, s1, feat1, out1);
  k_bias_relu<<<(NN * HD + 255) / 256, 256, 0, stream>>>(out1, b1);

  // layer 2
  k_gemm2<<<(MT + 3) / 4, 128, 0, stream>>>(out1, pw2, feat2);
  k_elr2<<<(NN + 255) / 256, 256, 0, stream>>>(feat2, al2, ar2, el2, er2);
  k_emax2<<<(EE + 255) / 256, 256, 0, stream>>>(src, dst, el2, er2, m2);
  k_esum2<<<(EE + 255) / 256, 256, 0, stream>>>(src, dst, el2, er2, m2, s2);
  k_eaggr2<<<(EE * NC + 255) / 256, 256, 0, stream>>>(src, dst, el2, er2, m2, s2, feat2, dout);
  k_bias2<<<(NN * NC + 255) / 256, 256, 0, stream>>>(dout, b2);
}
